// MultiHeadAttentionBlock_49667001811657
// MI455X (gfx1250) — matrix operand order repaired and checked by reading, not compiled, not run on hardware
//
#include <hip/hip_runtime.h>

#define D_MODEL 1024
#define NUM_HEADS 16
#define HEAD_DIM 64
#define BATCH 2
#define SEQ 2048
#define M_ROWS (BATCH * SEQ) /* 4096 */

typedef __attribute__((ext_vector_type(16))) _Float16 v16h;
typedef __attribute__((ext_vector_type(8)))  _Float16 v8h;
typedef __attribute__((ext_vector_type(8)))  float    v8f;

static __device__ __forceinline__ v8f zero_v8f() {
  v8f z;
#pragma unroll
  for (int i = 0; i < 8; ++i) z[i] = 0.0f;
  return z;
}

// Build a 16-half fragment from two 8-half (16B) chunks.
static __device__ __forceinline__ v16h load_frag_2x8(const _Float16* p0, const _Float16* p1) {
  v16h r;
  ((uint4*)&r)[0] = *(const uint4*)p0;
  ((uint4*)&r)[1] = *(const uint4*)p1;
  return r;
}

static __device__ __forceinline__ v8f wmma_f16(v16h a, v16h b, v8f c) {
  return __builtin_amdgcn_wmma_f32_16x16x32_f16(false, a, false, b, (short)0, c, false, false);
}

// ---------------- conversion kernels ----------------
__global__ void cvt_f32_to_f16(const float* __restrict__ in, _Float16* __restrict__ out, int n) {
  int i = blockIdx.x * blockDim.x + threadIdx.x;
  if (i < n) out[i] = (_Float16)in[i];
}

// w [K][N] f32 -> wT [N][K] f16 (so K is contiguous for B-fragments)
__global__ void cvt_transpose_w(const float* __restrict__ w, _Float16* __restrict__ wT,
                                int K, int N) {
  int i = blockIdx.x * blockDim.x + threadIdx.x; // n fastest -> coalesced reads
  int k = i / N, n = i % N;
  wT[(size_t)n * K + k] = (_Float16)w[(size_t)k * N + n];
}

// ---------------- WMMA GEMM: C = (A[M,K] x W[K,N] + bias) * outScale ----------------
// W supplied transposed: BT[N][K].  Tile 128(M) x 128(N), BK=32, 256 threads = 8 waves
// arranged 4(M) x 2(N); each wave computes 32x64 via 2x4 fragments (8 WMMA / K-step).
// Next K-tile is preloaded into registers before the WMMA block (software pipeline).
template <typename OutT>
__global__ __launch_bounds__(256)
void gemm_wmma(const _Float16* __restrict__ A, const _Float16* __restrict__ BT,
               const float* __restrict__ bias, OutT* __restrict__ C,
               int Mdim, int Ndim, int Kdim, float outScale) {
  __shared__ _Float16 As[128 * 40]; // padded stride 40 halves (80B) vs bank conflicts
  __shared__ _Float16 Bs[128 * 40];

  const int tid    = threadIdx.x;
  const int lane   = tid & 31;
  const int laneLo = lane & 15;
  const int hi     = lane >> 4;
  const int wid    = tid >> 5;
  const int waveM  = wid >> 1; // 0..3 -> 32 rows each
  const int waveN  = wid & 1;  // 0..1 -> 64 cols each

  const int blockN0 = blockIdx.x * 128;
  const int blockM0 = blockIdx.y * 128;

  // staging: each thread owns 16 halves of A-tile and 16 halves of B-tile
  const int sRow = tid >> 1;       // 0..127
  const int sCol = (tid & 1) * 16; // 0 / 16

  v8f acc[2][4];
#pragma unroll
  for (int mi = 0; mi < 2; ++mi)
#pragma unroll
    for (int ni = 0; ni < 4; ++ni) acc[mi][ni] = zero_v8f();

  const _Float16* gA = A + (size_t)(blockM0 + sRow) * Kdim + sCol;
  const _Float16* gB = BT + (size_t)(blockN0 + sRow) * Kdim + sCol;

  // preload first K-tile into registers
  uint4 ra0 = *(const uint4*)gA, ra1 = *(const uint4*)(gA + 8);
  uint4 rb0 = *(const uint4*)gB, rb1 = *(const uint4*)(gB + 8);

  for (int k0 = 0; k0 < Kdim; k0 += 32) {
    __syncthreads(); // previous tile's fragment reads complete
    *(uint4*)&As[sRow * 40 + sCol]     = ra0;
    *(uint4*)&As[sRow * 40 + sCol + 8] = ra1;
    *(uint4*)&Bs[sRow * 40 + sCol]     = rb0;
    *(uint4*)&Bs[sRow * 40 + sCol + 8] = rb1;
    __syncthreads(); // staging visible

    if (k0 + 32 < Kdim) { // issue next tile's global loads before the math
      gA += 32; gB += 32;
      ra0 = *(const uint4*)gA; ra1 = *(const uint4*)(gA + 8);
      rb0 = *(const uint4*)gB; rb1 = *(const uint4*)(gB + 8);
      __builtin_prefetch(gA + 32, 0, 1); // tile after next -> global_prefetch_b8
      __builtin_prefetch(gB + 32, 0, 1);
    }

    // A-frag (16x32 f16): lane m = laneLo, halves at k = 8*hi + {0..7} and +16
    v16h a[2], b[4];
#pragma unroll
    for (int mi = 0; mi < 2; ++mi) {
      const _Float16* p = &As[(waveM * 32 + mi * 16 + laneLo) * 40 + 8 * hi];
      a[mi] = load_frag_2x8(p, p + 16);
    }
    // B-frag (32x16): lane n = laneLo, halves at k = 8*hi + {0..7} and +16
#pragma unroll
    for (int ni = 0; ni < 4; ++ni) {
      const _Float16* p = &Bs[(waveN * 64 + ni * 16 + laneLo) * 40 + 8 * hi];
      b[ni] = load_frag_2x8(p, p + 16);
    }
#pragma unroll
    for (int mi = 0; mi < 2; ++mi)
#pragma unroll
      for (int ni = 0; ni < 4; ++ni)
        acc[mi][ni] = wmma_f16(a[mi], b[ni], acc[mi][ni]);
  }

  // epilogue: C layout -> lane holds (r + 8*hi, laneLo) per fragment
#pragma unroll
  for (int mi = 0; mi < 2; ++mi) {
#pragma unroll
    for (int ni = 0; ni < 4; ++ni) {
      const int n = blockN0 + waveN * 64 + ni * 16 + laneLo;
      const float bn = bias[n];
#pragma unroll
      for (int r = 0; r < 8; ++r) {
        const int m = blockM0 + waveM * 32 + mi * 16 + r + 8 * hi;
        C[(size_t)m * Ndim + n] = (OutT)((acc[mi][ni][r] + bn) * outScale);
      }
    }
  }
}

// ---------------- flash attention ----------------
// grid (SEQ/64, NUM_HEADS, BATCH), 128 threads (4 waves).
// Each wave owns 16 query rows; Dh=64 -> 4 output fragments per wave.
__global__ __launch_bounds__(128)
void flash_attn(const _Float16* __restrict__ Qp, const _Float16* __restrict__ Kp,
                const _Float16* __restrict__ Vp, _Float16* __restrict__ Op) {
  __shared__ _Float16 Kt[64 * 72];    // [key][dh], padded stride 72
  __shared__ _Float16 Vt[64 * 72];    // [dh][key] (transposed), padded stride 72
  __shared__ _Float16 Pw[4][16 * 72]; // per-wave probabilities, A-layout staging

  const int tid    = threadIdx.x;
  const int lane   = tid & 31;
  const int laneLo = lane & 15;
  const int hi     = lane >> 4;
  const int wave   = tid >> 5;

  const int mBlock = blockIdx.x;
  const int h      = blockIdx.y;
  const int b      = blockIdx.z;

  const size_t rowBase = (size_t)b * SEQ;
  const int colBase    = h * HEAD_DIM;

  // Q fragments for this wave's 16 rows (score scale folded into Q projection)
  v16h qa[2];
#pragma unroll
  for (int ks = 0; ks < 2; ++ks) {
    const _Float16* p = Qp + (rowBase + mBlock * 64 + wave * 16 + laneLo) * D_MODEL
                        + colBase + ks * 32 + 8 * hi;
    qa[ks] = load_frag_2x8(p, p + 16);
  }

  v8f of[4];
#pragma unroll
  for (int nf = 0; nf < 4; ++nf) of[nf] = zero_v8f();
  float m_i[8], l_i[8];
#pragma unroll
  for (int r = 0; r < 8; ++r) { m_i[r] = -3.0e38f; l_i[r] = 0.0f; }

  const int sKey = tid >> 1;       // 0..63
  const int sCol = (tid & 1) * 32; // 0 / 32 (32 halves per thread)

  for (int j = 0; j <= mBlock; ++j) { // causal: skip key blocks past the query tile
    const _Float16* gK = Kp + (rowBase + j * 64 + sKey) * D_MODEL + colBase + sCol;
    const _Float16* gV = Vp + (rowBase + j * 64 + sKey) * D_MODEL + colBase + sCol;
    if (j < mBlock) { // hint next block while this one computes
      __builtin_prefetch(gK + 64 * D_MODEL, 0, 1);
      __builtin_prefetch(gV + 64 * D_MODEL, 0, 1);
    }
    __syncthreads();
    { // stage K tile [key][dh]
#pragma unroll
      for (int c = 0; c < 4; ++c)
        *(uint4*)&Kt[sKey * 72 + sCol + c * 8] = *(const uint4*)(gK + c * 8);
    }
    { // stage V tile transposed -> Vt[dh][key]
#pragma unroll
      for (int c = 0; c < 4; ++c) {
        v8h t = *(const v8h*)(gV + c * 8);
#pragma unroll
        for (int e = 0; e < 8; ++e)
          Vt[(sCol + c * 8 + e) * 72 + sKey] = t[e];
      }
    }
    __syncthreads();

    // scores S = Q * K^T : contraction over Dh (2 k-steps of 32)
    v8f sf[4];
#pragma unroll
    for (int nf = 0; nf < 4; ++nf) sf[nf] = zero_v8f();
#pragma unroll
    for (int ks = 0; ks < 2; ++ks) {
#pragma unroll
      for (int nf = 0; nf < 4; ++nf) {
        const _Float16* p = &Kt[(nf * 16 + laneLo) * 72 + ks * 32 + 8 * hi];
        v16h kb = load_frag_2x8(p, p + 16);
        sf[nf] = wmma_f16(qa[ks], kb, sf[nf]);
      }
    }

    // causal mask only needed on the diagonal block (branchless selects keep EXEC full)
    if (j == mBlock) {
#pragma unroll
      for (int nf = 0; nf < 4; ++nf) {
        const int key = nf * 16 + laneLo;
#pragma unroll
        for (int r = 0; r < 8; ++r) {
          const int qrow = wave * 16 + r + 8 * hi;
          sf[nf][r] = (key > qrow) ? -1.0e30f : sf[nf][r];
        }
      }
    }

    // online softmax: rows live across lanes {0..15}/{16..31}; butterfly within halves
    float bm[8];
#pragma unroll
    for (int r = 0; r < 8; ++r) {
      float m0 = fmaxf(sf[0][r], sf[1][r]);
      float m1 = fmaxf(sf[2][r], sf[3][r]);
      bm[r] = fmaxf(m0, m1);
    }
#pragma unroll
    for (int off = 8; off >= 1; off >>= 1)
#pragma unroll
      for (int r = 0; r < 8; ++r)
        bm[r] = fmaxf(bm[r], __shfl_xor(bm[r], off, 32));

    float ps[8];
#pragma unroll
    for (int r = 0; r < 8; ++r) {
      const float newm  = fmaxf(m_i[r], bm[r]);
      const float alpha = __expf(m_i[r] - newm);
      m_i[r] = newm;
      float s = 0.0f;
#pragma unroll
      for (int nf = 0; nf < 4; ++nf) {
        float p = __expf(sf[nf][r] - newm);
        sf[nf][r] = p;
        s += p;
      }
      ps[r] = s;
      l_i[r] *= alpha;
#pragma unroll
      for (int nf = 0; nf < 4; ++nf) of[nf][r] *= alpha;
    }
#pragma unroll
    for (int off = 8; off >= 1; off >>= 1)
#pragma unroll
      for (int r = 0; r < 8; ++r)
        ps[r] += __shfl_xor(ps[r], off, 32);
#pragma unroll
    for (int r = 0; r < 8; ++r) l_i[r] += ps[r];

    // re-shape P from C-layout to A-layout through per-wave LDS
#pragma unroll
    for (int nf = 0; nf < 4; ++nf)
#pragma unroll
      for (int r = 0; r < 8; ++r)
        Pw[wave][(r + 8 * hi) * 72 + nf * 16 + laneLo] = (_Float16)sf[nf][r];
    __syncthreads();

    // O += P * V (contraction over 64 keys, 2 k-steps)
#pragma unroll
    for (int ks = 0; ks < 2; ++ks) {
      const _Float16* pp = &Pw[wave][laneLo * 72 + ks * 32 + 8 * hi];
      v16h pa = load_frag_2x8(pp, pp + 16);
#pragma unroll
      for (int nf = 0; nf < 4; ++nf) {
        const _Float16* vp = &Vt[(nf * 16 + laneLo) * 72 + ks * 32 + 8 * hi];
        v16h vb = load_frag_2x8(vp, vp + 16);
        of[nf] = wmma_f16(pa, vb, of[nf]);
      }
    }
  }

  // normalize and store (heads concatenated via column offset)
#pragma unroll
  for (int r = 0; r < 8; ++r) {
    const float inv  = 1.0f / l_i[r];
    const size_t row = rowBase + mBlock * 64 + wave * 16 + r + 8 * hi;
#pragma unroll
    for (int nf = 0; nf < 4; ++nf)
      Op[row * D_MODEL + colBase + nf * 16 + laneLo] = (_Float16)(of[nf][r] * inv);
  }
}

extern "C" void kernel_launch(void* const* d_in, const int* in_sizes, int n_in,
                              void* d_out, int out_size, void* d_ws, size_t ws_size,
                              hipStream_t stream) {
  (void)in_sizes; (void)n_in; (void)out_size; (void)ws_size;
  const float* q   = (const float*)d_in[0];
  const float* k   = (const float*)d_in[1];
  const float* v   = (const float*)d_in[2];
  /* d_in[3] = mask: causal, handled analytically */
  const float* w_q = (const float*)d_in[4];
  const float* b_q = (const float*)d_in[5];
  const float* w_k = (const float*)d_in[6];
  const float* b_k = (const float*)d_in[7];
  const float* w_v = (const float*)d_in[8];
  const float* b_v = (const float*)d_in[9];
  const float* w_o = (const float*)d_in[10];
  const float* b_o = (const float*)d_in[11];
  float* out = (float*)d_out;

  const size_t ACT = (size_t)M_ROWS * D_MODEL;   // 4M elems
  const size_t WEL = (size_t)D_MODEL * D_MODEL;  // 1M elems
  _Float16* ws  = (_Float16*)d_ws;
  _Float16* qc  = ws;          _Float16* kc  = qc + ACT;  _Float16* vc  = kc + ACT;
  _Float16* wqT = vc + ACT;    _Float16* wkT = wqT + WEL;
  _Float16* wvT = wkT + WEL;   _Float16* woT = wvT + WEL;
  _Float16* Qp  = woT + WEL;   _Float16* Kp  = Qp + ACT;  _Float16* Vp  = Kp + ACT;
  _Float16* aO  = Vp + ACT;    // total: 64 MiB of workspace

  cvt_f32_to_f16<<<(int)(ACT / 256), 256, 0, stream>>>(q, qc, (int)ACT);
  cvt_f32_to_f16<<<(int)(ACT / 256), 256, 0, stream>>>(k, kc, (int)ACT);
  cvt_f32_to_f16<<<(int)(ACT / 256), 256, 0, stream>>>(v, vc, (int)ACT);
  cvt_transpose_w<<<(int)(WEL / 256), 256, 0, stream>>>(w_q, wqT, D_MODEL, D_MODEL);
  cvt_transpose_w<<<(int)(WEL / 256), 256, 0, stream>>>(w_k, wkT, D_MODEL, D_MODEL);
  cvt_transpose_w<<<(int)(WEL / 256), 256, 0, stream>>>(w_v, wvT, D_MODEL, D_MODEL);
  cvt_transpose_w<<<(int)(WEL / 256), 256, 0, stream>>>(w_o, woT, D_MODEL, D_MODEL);

  dim3 ggrid(D_MODEL / 128, M_ROWS / 128);
  const float qscale = 0.125f; // 1/sqrt(HEAD_DIM), folded into Q projection
  gemm_wmma<_Float16><<<ggrid, 256, 0, stream>>>(qc, wqT, b_q, Qp, M_ROWS, D_MODEL, D_MODEL, qscale);
  gemm_wmma<_Float16><<<ggrid, 256, 0, stream>>>(kc, wkT, b_k, Kp, M_ROWS, D_MODEL, D_MODEL, 1.0f);
  gemm_wmma<_Float16><<<ggrid, 256, 0, stream>>>(vc, wvT, b_v, Vp, M_ROWS, D_MODEL, D_MODEL, 1.0f);

  dim3 agrid(SEQ / 64, NUM_HEADS, BATCH);
  flash_attn<<<agrid, 128, 0, stream>>>(Qp, Kp, Vp, aO);

  gemm_wmma<float><<<ggrid, 256, 0, stream>>>(aO, woT, b_o, out, M_ROWS, D_MODEL, D_MODEL, 1.0f);
}
